// MS_DefCorrelation_Module_45664092291264
// MI455X (gfx1250) — matrix operand order repaired and checked by reading, not compiled, not run on hardware
//
#include <hip/hip_runtime.h>
#include <hip/hip_bf16.h>

typedef __attribute__((ext_vector_type(16))) _Float16 v16h;
typedef __attribute__((ext_vector_type(8)))  float    v8f;
typedef __attribute__((ext_vector_type(2)))  _Float16 h2;

#define B_  2
#define C_  64
#define T_  16
#define H_  64
#define W_  64
#define HW_ 4096
#define BT_ 32
#define R_  4

struct alignas(16) H8 { _Float16 v[8]; };

__device__ __forceinline__ float sigmoidf_(float v) { return 1.f / (1.f + __expf(-v)); }

// ---------------- bilinear sampling helper (matches reference grid_sample) ---
struct Bil { int x0c, x1c, y0c, y1c; float w00, w01, w10, w11; };

__device__ __forceinline__ Bil make_bil(float ox, float oy, int wi, int hi) {
  float gnx = 2.f * wi / (W_ - 1) - 1.f + ox / (W_ - 1);
  float gny = 2.f * hi / (H_ - 1) - 1.f + oy / (H_ - 1);
  float gx = ((gnx + 1.f) * W_ - 1.f) * 0.5f;
  float gy = ((gny + 1.f) * H_ - 1.f) * 0.5f;
  float fx = floorf(gx), fy = floorf(gy);
  float wx = gx - fx, wy = gy - fy;
  int x0 = (int)fx, y0 = (int)fy, x1 = x0 + 1, y1 = y0 + 1;
  float vx0 = (x0 >= 0 && x0 < W_) ? 1.f : 0.f;
  float vx1 = (x1 >= 0 && x1 < W_) ? 1.f : 0.f;
  float vy0 = (y0 >= 0 && y0 < H_) ? 1.f : 0.f;
  float vy1 = (y1 >= 0 && y1 < H_) ? 1.f : 0.f;
  Bil r;
  r.w00 = (1.f - wx) * (1.f - wy) * vx0 * vy0;
  r.w01 = wx * (1.f - wy) * vx1 * vy0;
  r.w10 = (1.f - wx) * wy * vx0 * vy1;
  r.w11 = wx * wy * vx1 * vy1;
  r.x0c = min(max(x0, 0), W_ - 1); r.x1c = min(max(x1, 0), W_ - 1);
  r.y0c = min(max(y0, 0), H_ - 1); r.y1c = min(max(y1, 0), H_ - 1);
  return r;
}

__device__ __forceinline__ float bsample(const float* __restrict__ im, const Bil& b) {
  return b.w00 * im[b.y0c * W_ + b.x0c] + b.w01 * im[b.y0c * W_ + b.x1c]
       + b.w10 * im[b.y1c * W_ + b.x0c] + b.w11 * im[b.y1c * W_ + b.x1c];
}

// ---------------- W-repack kernels: fp32 OIHW -> f16 chunk-major ------------
// w1p[((kc*64 + n)*32) + kl] = w1[(n*128 + (kc&3)*32 + kl)*9 + (kc>>2)]
__global__ void __launch_bounds__(256) k_repack_w1(const float* __restrict__ w1,
                                                   _Float16* __restrict__ w1p) {
  int i  = blockIdx.x * 256 + threadIdx.x;       // 36*64*32 = 73728
  int kl = i & 31;
  int n  = (i >> 5) & 63;
  int kc = i >> 11;
  int tap = kc >> 2, c0 = (kc & 3) * 32;
  w1p[i] = (_Float16)w1[(n * 128 + c0 + kl) * 9 + tap];
}

// w2p[((kc*16 + n)*32) + kl] : n>=6 zero-padded
__global__ void __launch_bounds__(256) k_repack_w2(const float* __restrict__ w2,
                                                   _Float16* __restrict__ w2p) {
  int i  = blockIdx.x * 256 + threadIdx.x;       // 18*16*32 = 9216
  int kl = i & 31;
  int n  = (i >> 5) & 15;
  int kc = i >> 9;
  int tap = kc >> 1, c0 = (kc & 1) * 32;
  w2p[i] = (n < 6) ? (_Float16)w2[(n * 64 + c0 + kl) * 9 + tap] : (_Float16)0.f;
}

// ---------------- K1: 1x1x1 down conv, C=64 -> r=4 --------------------------
__global__ void __launch_bounds__(256) k_down(const float* __restrict__ x,
                                              const float* __restrict__ dw,
                                              float* __restrict__ xagg) {
  int i  = blockIdx.x * 256 + threadIdx.x;   // over B*T*HW = 131072
  int hw = i & (HW_ - 1);
  int t  = (i >> 12) & (T_ - 1);
  int b  = i >> 16;
  float acc[R_] = {0.f, 0.f, 0.f, 0.f};
  for (int c = 0; c < C_; ++c) {
    float xv = x[((size_t)((b * C_ + c) * T_ + t)) * HW_ + hw];
#pragma unroll
    for (int j = 0; j < R_; ++j) acc[j] += xv * dw[j * C_ + c];
  }
#pragma unroll
  for (int j = 0; j < R_; ++j)
    xagg[((size_t)((b * R_ + j) * T_ + t)) * HW_ + hw] = acc[j];
}

// ---------------- K2: 3-scale grouped 9x3x3 dilated convs + blend -----------
__global__ void __launch_bounds__(256) k_agg(const float* __restrict__ xagg,
    const float* __restrict__ w1, const float* __restrict__ b1,
    const float* __restrict__ w2, const float* __restrict__ b2,
    const float* __restrict__ w3, const float* __restrict__ b3,
    const float* __restrict__ aw, float* __restrict__ agg) {
  int i  = blockIdx.x * 256 + threadIdx.x;   // over B*R*T*HW = 524288
  int hw = i & 4095; int w = hw & 63; int h = hw >> 6;
  int t  = (i >> 12) & 15;
  int j  = (i >> 16) & 3;
  int b  = i >> 18;
  const float* ws[3] = {w1, w2, w3};
  const float* bs[3] = {b1, b2, b3};
  float tot = 0.f;
  for (int s = 0; s < 3; ++s) {
    int d = s + 1;
    float acc = bs[s][j];
    const float* wp = ws[s] + j * 81;
    for (int kt = 0; kt < 9; ++kt) {
      int tt = t + kt - 4;
      if (tt < 0 || tt >= T_) continue;
      const float* src = xagg + ((size_t)((b * R_ + j) * T_ + tt)) * HW_;
      for (int kh = 0; kh < 3; ++kh) {
        int hh = h + d * (kh - 1);
        if (hh < 0 || hh >= H_) continue;
        for (int kw = 0; kw < 3; ++kw) {
          int ww = w + d * (kw - 1);
          if (ww < 0 || ww >= W_) continue;
          acc += src[hh * W_ + ww] * wp[kt * 9 + kh * 3 + kw];
        }
      }
    }
    tot += acc * aw[s];
  }
  agg[i] = tot;
}

// ---------------- K3: WMMA implicit-GEMM conv  (Cin=128 -> Cout=64, 3x3) ----
// One workgroup = one (frame, output row). M=64 pixels, N=64 out-ch, K=1152.
// K tap-major: k = tap*128 + c, 36 chunks of 32. Weights pre-packed f16.
__global__ void __launch_bounds__(256) k_conv1_wmma(const float* __restrict__ x,
    const _Float16* __restrict__ w1p, const float* __restrict__ b1,
    _Float16* __restrict__ hout, int dir) {
  const int f = blockIdx.x >> 6;       // frame 0..31
  const int y = blockIdx.x & 63;       // output row
  const int b = f >> 4;
  const int t = f & 15;
  int t2 = t + dir; t2 = t2 < 0 ? 0 : (t2 > T_ - 1 ? T_ - 1 : t2);

  __shared__ _Float16 sIn[3][66][132];   // [dy][x+dx][c]  (c padded 128->132)
  __shared__ _Float16 sB[64][40];        // [n][k_local]   (32->40: 80B rows)

  const int tid = threadIdx.x;

  // Zero the two always-OOB boundary columns (gx = -1 and gx = 64).
  for (int i = tid; i < 3 * 128 * 2; i += 256) {
    int r  = i & 255;
    int dy = i >> 8;
    sIn[dy][(r & 1) ? 65 : 0][r >> 1] = (_Float16)0.f;
  }
  // Stage interior: float4 loads of channel pairs -> packed h2 LDS stores.
  for (int i = tid; i < 3 * 64 * 16; i += 256) {   // (dy, cpair, quad)
    int q  = i & 15;
    int cp = (i >> 4) & 63;
    int dy = i >> 10;
    int c  = cp * 2;
    int gy = y + dy - 1;
    float4 v0 = make_float4(0.f, 0.f, 0.f, 0.f), v1 = v0;
    if (gy >= 0 && gy < H_) {
      int tt = (c < 64) ? t : t2;
      int cc = c & 63;
      const float* base = x + ((size_t)((b * C_ + cc) * T_ + tt)) * HW_ + gy * W_ + q * 4;
      v0 = *(const float4*)base;
      v1 = *(const float4*)(base + (size_t)T_ * HW_ * 0 + ((size_t)HW_ * T_));  // next channel plane
    }
    int col = 1 + q * 4;
    h2 p;
    p.x = (_Float16)v0.x; p.y = (_Float16)v1.x; *(h2*)&sIn[dy][col + 0][c] = p;
    p.x = (_Float16)v0.y; p.y = (_Float16)v1.y; *(h2*)&sIn[dy][col + 1][c] = p;
    p.x = (_Float16)v0.z; p.y = (_Float16)v1.z; *(h2*)&sIn[dy][col + 2][c] = p;
    p.x = (_Float16)v0.w; p.y = (_Float16)v1.w; *(h2*)&sIn[dy][col + 3][c] = p;
  }

  const int wave  = tid >> 5;
  const int lane  = tid & 31;
  const int m_sub = wave & 3;             // pixel subtile 0..3
  const int n_b   = (wave >> 2) * 2;      // out-channel subtile pair base
  const int lrow  = lane & 15;
  const int lhi   = lane >> 4;

  // Per-chunk weight staging: one 16B load + one 16B LDS store per thread.
  const int sg_n   = tid >> 2;            // 0..63
  const int sg_kl0 = (tid & 3) * 8;       // 0,8,16,24

  v8f acc0 = {};
  v8f acc1 = {};

  for (int kc = 0; kc < 36; ++kc) {
    const int tap = kc >> 2;              // 0..8
    const int c0  = (kc & 3) * 32;        // channel block
    const int dy  = tap / 3;
    const int dx  = tap % 3;
    __syncthreads();
    *(H8*)&sB[sg_n][sg_kl0] = *(const H8*)&w1p[((size_t)kc * 64 + sg_n) * 32 + sg_kl0];
    __syncthreads();

    // A fragment (ISA 16-bit A 16x32 layout)
    v16h a;
    {
      int m = m_sub * 16 + lrow;
      const _Float16* base = &sIn[dy][m + dx][c0 + lhi * 8];
#pragma unroll
      for (int p = 0; p < 8; ++p) {
        int kk = ((p & 4) ? 16 : 0) + (p & 3) * 2;
        h2 v = *(const h2*)(base + kk);
        a[2 * p] = v.x; a[2 * p + 1] = v.y;
      }
    }
    // B fragments (ISA 16-bit B 32x16): lanes 0-15 K=0..7,16..23, lanes 16-31 K=8..15,24..31
    v16h bf0, bf1;
    {
      int kb = lhi * 8;
      const _Float16* r0 = &sB[(n_b    ) * 16 + lrow][kb];
      const _Float16* r1 = &sB[(n_b + 1) * 16 + lrow][kb];
#pragma unroll
      for (int p = 0; p < 8; ++p) {
        h2 v0 = *(const h2*)(r0 + ((p & 4) ? 16 : 0) + (p & 3) * 2);
        h2 v1 = *(const h2*)(r1 + ((p & 4) ? 16 : 0) + (p & 3) * 2);
        bf0[2 * p] = v0.x; bf0[2 * p + 1] = v0.y;
        bf1[2 * p] = v1.x; bf1[2 * p + 1] = v1.y;
      }
    }
    acc0 = __builtin_amdgcn_wmma_f32_16x16x32_f16(false, a, false, bf0, (short)0, acc0, false, false);
    acc1 = __builtin_amdgcn_wmma_f32_16x16x32_f16(false, a, false, bf1, (short)0, acc1, false, false);
  }

  // Epilogue: bias + ReLU, pack 8 contiguous f16 -> one 16B store per lane/tile.
  const int x0 = m_sub * 16 + lhi * 8;    // D layout: M = vgpr + 8*lhi
#pragma unroll
  for (int tile = 0; tile < 2; ++tile) {
    int n = (n_b + tile) * 16 + lrow;
    float bias = b1[n];
    H8 pack;
#pragma unroll
    for (int vi = 0; vi < 8; ++vi) {
      float val = (tile == 0 ? acc0[vi] : acc1[vi]) + bias;
      pack.v[vi] = (_Float16)(val > 0.f ? val : 0.f);
    }
    *(H8*)&hout[(((size_t)f * 64 + n) * H_ + y) * W_ + x0] = pack;
  }
}

// ---------------- K4: WMMA 3x3 conv 64 -> 6 (offset head, N padded to 16) ---
// One workgroup (128 thr, 4 waves) = one (frame, row). M=64, N=16, K=576.
__global__ void __launch_bounds__(128) k_conv2_wmma(const _Float16* __restrict__ hin,
    const _Float16* __restrict__ w2p, const float* __restrict__ b2,
    float* __restrict__ off) {
  const int f = blockIdx.x >> 6;
  const int y = blockIdx.x & 63;

  __shared__ _Float16 sIn2[3][66][68];   // [dy][x+dx][c]  (64->68 pad)
  __shared__ _Float16 sB2[18][16][40];   // all 18 K-chunks resident

  const int tid = threadIdx.x;

  // Stage all weights once: 9216 halfs, 16B vector copies.
  for (int i = tid; i < 18 * 16 * 4; i += 128) {
    int kl0 = (i & 3) * 8;
    int n   = (i >> 2) & 15;
    int kc  = i >> 6;
    *(H8*)&sB2[kc][n][kl0] = *(const H8*)&w2p[((size_t)kc * 16 + n) * 32 + kl0];
  }
  // Zero boundary columns.
  for (int i = tid; i < 3 * 64 * 2; i += 128) {
    int r  = i & 127;
    int dy = i >> 7;
    sIn2[dy][(r & 1) ? 65 : 0][r >> 1] = (_Float16)0.f;
  }
  // Stage interior from f16 hidden buffer: channel-pair H8 loads -> h2 stores.
  for (int i = tid; i < 3 * 32 * 8; i += 128) {    // (dy, cpair, oct)
    int q  = i & 7;
    int cp = (i >> 3) & 31;
    int dy = i >> 8;
    int c  = cp * 2;
    int gy = y + dy - 1;
    H8 v0, v1;
#pragma unroll
    for (int j = 0; j < 8; ++j) { v0.v[j] = (_Float16)0.f; v1.v[j] = (_Float16)0.f; }
    if (gy >= 0 && gy < H_) {
      const _Float16* base = hin + ((size_t)(f * 64 + c)) * HW_ + gy * W_ + q * 8;
      v0 = *(const H8*)base;
      v1 = *(const H8*)(base + HW_);
    }
    int col = 1 + q * 8;
#pragma unroll
    for (int j = 0; j < 8; ++j) {
      h2 p; p.x = v0.v[j]; p.y = v1.v[j];
      *(h2*)&sIn2[dy][col + j][c] = p;
    }
  }
  __syncthreads();

  const int wave  = tid >> 5;             // = m_sub 0..3
  const int lane  = tid & 31;
  const int lrow  = lane & 15;
  const int lhi   = lane >> 4;

  v8f acc = {};
  for (int kc = 0; kc < 18; ++kc) {
    const int tap = kc >> 1;
    const int c0  = (kc & 1) * 32;
    const int dy  = tap / 3;
    const int dx  = tap % 3;
    v16h a;
    {
      int m = wave * 16 + lrow;
      const _Float16* base = &sIn2[dy][m + dx][c0 + lhi * 8];
#pragma unroll
      for (int p = 0; p < 8; ++p) {
        int kk = ((p & 4) ? 16 : 0) + (p & 3) * 2;
        h2 v = *(const h2*)(base + kk);
        a[2 * p] = v.x; a[2 * p + 1] = v.y;
      }
    }
    v16h bf;
    {
      int kb = lhi * 8;
      const _Float16* r0 = &sB2[kc][lrow][kb];
#pragma unroll
      for (int p = 0; p < 8; ++p) {
        h2 v0 = *(const h2*)(r0 + ((p & 4) ? 16 : 0) + (p & 3) * 2);
        bf[2 * p] = v0.x; bf[2 * p + 1] = v0.y;
      }
    }
    acc = __builtin_amdgcn_wmma_f32_16x16x32_f16(false, a, false, bf, (short)0, acc, false, false);
  }

  // Epilogue: only n<6 are real output channels.
  if (lrow < 6) {
    float bias = b2[lrow];
    int x0 = wave * 16 + lhi * 8;
    float* dst = off + ((size_t)(f * 6 + lrow)) * HW_ + y * W_ + x0;
    float4 lo = make_float4(acc[0] + bias, acc[1] + bias, acc[2] + bias, acc[3] + bias);
    float4 hi = make_float4(acc[4] + bias, acc[5] + bias, acc[6] + bias, acc[7] + bias);
    *(float4*)dst = lo;
    *(float4*)(dst + 4) = hi;
  }
}

// ---------------- K5: deformable correlation -> per-pixel (avg, max, var) ---
__global__ void __launch_bounds__(256) k_corr(const float* __restrict__ x,
    const float* __restrict__ off, int dir, float* __restrict__ stats) {
  int i  = blockIdx.x * 256 + threadIdx.x;   // over BT*HW
  int hw = i & 4095; int w = hw & 63; int h = hw >> 6;
  int f  = i >> 12;
  int b  = f >> 4, t = f & 15;
  int t2 = t + dir; t2 = t2 < 0 ? 0 : (t2 > T_ - 1 ? T_ - 1 : t2);
  Bil bl[3];
#pragma unroll
  for (int s = 0; s < 3; ++s)
    bl[s] = make_bil(off[((size_t)(f * 6 + 2 * s)) * HW_ + hw],
                     off[((size_t)(f * 6 + 2 * s + 1)) * HW_ + hw], w, h);
  float dot[3] = {0.f, 0.f, 0.f};
  for (int c = 0; c < 64; ++c) {
    const float* imx = x + ((size_t)((b * C_ + c) * T_ + t )) * HW_;
    const float* imy = x + ((size_t)((b * C_ + c) * T_ + t2)) * HW_;
    float xv = imx[hw];
#pragma unroll
    for (int s = 0; s < 3; ++s) dot[s] += xv * bsample(imy, bl[s]);
  }
  float c0 = dot[0] / 64.f, c1 = dot[1] / 64.f, c2 = dot[2] / 64.f;
  float avg = (c0 + c1 + c2) * (1.f / 3.f);
  float mx  = fmaxf(c0, fmaxf(c1, c2));
  float var = ((c0 - avg) * (c0 - avg) + (c1 - avg) * (c1 - avg) + (c2 - avg) * (c2 - avg)) * 0.5f; // ddof=1
  stats[((size_t)(f * 3 + 0)) * HW_ + hw] = avg;
  stats[((size_t)(f * 3 + 1)) * HW_ + hw] = mx;
  stats[((size_t)(f * 3 + 2)) * HW_ + hw] = var;
}

// ---------------- K6: corr_filter attention -> sigmoid(attn*avg)-0.5 --------
__global__ void __launch_bounds__(256) k_attn(const float* __restrict__ stats,
    const float* __restrict__ cw1, const float* __restrict__ cw2,
    float* __restrict__ attn) {
  int i  = blockIdx.x * 256 + threadIdx.x;
  int hw = i & 4095; int w = hw & 63; int h = hw >> 6;
  int f  = i >> 12;
  const float* pa = stats + ((size_t)(f * 3 + 0)) * HW_;
  const float* pm = stats + ((size_t)(f * 3 + 1)) * HW_;
  const float* pv = stats + ((size_t)(f * 3 + 2)) * HW_;
  float acc = 0.f;
  for (int ky = 0; ky < 3; ++ky) {
    int gy = h + ky - 1; if (gy < 0 || gy >= H_) continue;
    for (int kx = 0; kx < 3; ++kx) {
      int gx = w + kx - 1; if (gx < 0 || gx >= W_) continue;
      int o = gy * W_ + gx;
      acc += cw1[ky * 3 + kx] * pa[o] + cw1[9 + ky * 3 + kx] * pm[o] + cw2[ky * 3 + kx] * pv[o];
    }
  }
  float am = sigmoidf_(acc);
  float al = am * pa[hw];
  attn[i] = sigmoidf_(al) - 0.5f;
}

// ---------------- K7: warp + fuse + long-term gate -> output ----------------
__global__ void __launch_bounds__(256) k_final(const float* __restrict__ x,
    const float* __restrict__ agg, const float* __restrict__ back_w,
    const float* __restrict__ off_l, const float* __restrict__ off_r,
    const float* __restrict__ attn_l, const float* __restrict__ attn_r,
    const float* __restrict__ fw, float* __restrict__ out) {
  int i  = blockIdx.x * 256 + threadIdx.x;   // over B*T*HW
  int hw = i & 4095; int w = hw & 63; int h = hw >> 6;
  int t  = (i >> 12) & 15;
  int b  = i >> 16;
  int f  = b * T_ + t;
  int tl = min(t + 1, T_ - 1), tr = max(t - 1, 0);
  Bil bL = make_bil(off_l[((size_t)(f * 6 + 0)) * HW_ + hw],
                    off_l[((size_t)(f * 6 + 1)) * HW_ + hw], w, h);
  Bil bR = make_bil(off_r[((size_t)(f * 6 + 0)) * HW_ + hw],
                    off_r[((size_t)(f * 6 + 1)) * HW_ + hw], w, h);
  float sal = attn_l[f * HW_ + hw];
  float sar = attn_r[f * HW_ + hw];
  float a0 = agg[((size_t)((b * R_ + 0) * T_ + t)) * HW_ + hw];
  float a1 = agg[((size_t)((b * R_ + 1) * T_ + t)) * HW_ + hw];
  float a2 = agg[((size_t)((b * R_ + 2) * T_ + t)) * HW_ + hw];
  float a3 = agg[((size_t)((b * R_ + 3) * T_ + t)) * HW_ + hw];
  float f0 = fw[0], f1 = fw[1];
  for (int c = 0; c < C_; ++c) {
    const float* imL = x + ((size_t)((b * C_ + c) * T_ + tl)) * HW_;
    const float* imR = x + ((size_t)((b * C_ + c) * T_ + tr)) * HW_;
    float wl = bsample(imL, bL);
    float wr = bsample(imR, bR);
    float lt = sigmoidf_(back_w[c * 4 + 0] * a0 + back_w[c * 4 + 1] * a1 +
                         back_w[c * 4 + 2] * a2 + back_w[c * 4 + 3] * a3) - 0.5f;
    out[((size_t)((b * C_ + c) * T_ + t)) * HW_ + hw] = (f0 * wl * sal + f1 * wr * sar) * lt;
  }
}

// ---------------------------------------------------------------------------
extern "C" void kernel_launch(void* const* d_in, const int* in_sizes, int n_in,
                              void* d_out, int out_size, void* d_ws, size_t ws_size,
                              hipStream_t stream) {
  (void)in_sizes; (void)n_in; (void)out_size; (void)ws_size;
  const float* x        = (const float*)d_in[0];
  const float* ol_w1    = (const float*)d_in[1];
  const float* ol_b1    = (const float*)d_in[2];
  const float* ol_w2    = (const float*)d_in[3];
  const float* ol_b2    = (const float*)d_in[4];
  const float* or_w1    = (const float*)d_in[5];
  const float* or_b1    = (const float*)d_in[6];
  const float* or_w2    = (const float*)d_in[7];
  const float* or_b2    = (const float*)d_in[8];
  const float* cf_w1    = (const float*)d_in[9];
  const float* cf_w2    = (const float*)d_in[10];
  const float* down_w   = (const float*)d_in[11];
  const float* sa1_w    = (const float*)d_in[12];
  const float* sa1_b    = (const float*)d_in[13];
  const float* sa2_w    = (const float*)d_in[14];
  const float* sa2_b    = (const float*)d_in[15];
  const float* sa3_w    = (const float*)d_in[16];
  const float* sa3_b    = (const float*)d_in[17];
  const float* agg_w    = (const float*)d_in[18];
  const float* back_w   = (const float*)d_in[19];
  const float* fusion_w = (const float*)d_in[20];
  float* out = (float*)d_out;

  char* ws = (char*)d_ws;                            // ~50 MB layout
  float*    xagg  = (float*)(ws);                    // 2 MB
  float*    agg   = (float*)(ws + (2u  << 20));      // 2 MB
  _Float16* h_l   = (_Float16*)(ws + (4u  << 20));   // 16 MB
  _Float16* h_r   = (_Float16*)(ws + (20u << 20));   // 16 MB
  float*    off_l = (float*)(ws + (36u << 20));      // 3 MB
  float*    off_r = (float*)(ws + (39u << 20));      // 3 MB
  float*    st_l  = (float*)(ws + (42u << 20));      // 1.5 MB (2 MB slot)
  float*    st_r  = (float*)(ws + (44u << 20));      // 1.5 MB (2 MB slot)
  float*    at_l  = (float*)(ws + (46u << 20));      // 0.5 MB (1 MB slot)
  float*    at_r  = (float*)(ws + (47u << 20));      // 0.5 MB
  _Float16* w1p_l = (_Float16*)(ws + (48u << 20));              // 144 KB (512K slot)
  _Float16* w1p_r = (_Float16*)(ws + (48u << 20) + (512u<<10)); // 144 KB
  _Float16* w2p_l = (_Float16*)(ws + (49u << 20));              // 18 KB (64K slot)
  _Float16* w2p_r = (_Float16*)(ws + (49u << 20) + (64u<<10));  // 18 KB

  k_repack_w1<<<288, 256, 0, stream>>>(ol_w1, w1p_l);
  k_repack_w1<<<288, 256, 0, stream>>>(or_w1, w1p_r);
  k_repack_w2<<<36,  256, 0, stream>>>(ol_w2, w2p_l);
  k_repack_w2<<<36,  256, 0, stream>>>(or_w2, w2p_r);

  k_down <<<512,  256, 0, stream>>>(x, down_w, xagg);
  k_agg  <<<2048, 256, 0, stream>>>(xagg, sa1_w, sa1_b, sa2_w, sa2_b,
                                    sa3_w, sa3_b, agg_w, agg);
  k_conv1_wmma<<<2048, 256, 0, stream>>>(x, w1p_l, ol_b1, h_l, +1);
  k_conv1_wmma<<<2048, 256, 0, stream>>>(x, w1p_r, or_b1, h_r, -1);
  k_conv2_wmma<<<2048, 128, 0, stream>>>(h_l, w2p_l, ol_b2, off_l);
  k_conv2_wmma<<<2048, 128, 0, stream>>>(h_r, w2p_r, or_b2, off_r);
  k_corr <<<512, 256, 0, stream>>>(x, off_l, +1, st_l);
  k_corr <<<512, 256, 0, stream>>>(x, off_r, -1, st_r);
  k_attn <<<512, 256, 0, stream>>>(st_l, cf_w1, cf_w2, at_l);
  k_attn <<<512, 256, 0, stream>>>(st_r, cf_w1, cf_w2, at_r);
  k_final<<<512, 256, 0, stream>>>(x, agg, back_w, off_l, off_r,
                                   at_l, at_r, fusion_w, out);
}
